// Retention_37477884625677
// MI455X (gfx1250) — matrix operand order repaired and checked by reading, not compiled, not run on hardware
//
#include <hip/hip_runtime.h>
#include <hip/hip_bf16.h>
#include <math.h>

#define GAMMA_F 0.9865f
#define NB 8
#define NS 2048
#define ND 128
#define BSD (8 * 2048 * 128)
#define RSQRT_D 0.08838834764831845f

typedef __bf16 bf16_t;
typedef __attribute__((ext_vector_type(16))) __bf16 v16bf;
typedef __attribute__((ext_vector_type(8)))  __bf16 v8bf;
typedef __attribute__((ext_vector_type(8)))  float  v8f;

__device__ __forceinline__ v8f wmma_bf16(v16bf a, v16bf b, v8f c) {
  // (neg_a, A, neg_b, B, c_mod, C, reuse_a, reuse_b)
  return __builtin_amdgcn_wmma_f32_16x16x32_bf16(false, a, false, b, (short)0, c,
                                                 false, false);
}

// A fragment (16x32 bf16, row-major [M][K] in LDS):
// lane: M = lane&15, h = lane>>4; VGPR0-3 hold K = 8h..8h+7, VGPR4-7 hold K = 16+8h..16+8h+7.
__device__ __forceinline__ v16bf load_frag_a(const bf16_t* base, int stride, int lane) {
  const bf16_t* p = base + (lane & 15) * stride + 8 * (lane >> 4);
  v8bf lo = *(const v8bf*)(p);
  v8bf hi = *(const v8bf*)(p + 16);
  return __builtin_shufflevector(lo, hi, 0, 1, 2, 3, 4, 5, 6, 7,
                                         8, 9, 10, 11, 12, 13, 14, 15);
}

// B fragment (32x16 bf16) loaded from row-major [N][K] in LDS:
// lane: N = lane&15, h = lane>>4; VGPR0-3 hold K = 8h..8h+7, VGPR4-7 hold K = 16+8h..16+8h+7.
__device__ __forceinline__ v16bf load_frag_b(const bf16_t* base, int stride, int lane) {
  const bf16_t* p = base + (lane & 15) * stride + 8 * (lane >> 4);
  v8bf lo = *(const v8bf*)(p);
  v8bf hi = *(const v8bf*)(p + 16);
  return __builtin_shufflevector(lo, hi, 0, 1, 2, 3, 4, 5, 6, 7,
                                         8, 9, 10, 11, 12, 13, 14, 15);
}

// ---------------------------------------------------------------------------
// Stage A: Y = X @ W   (per matrix; [2048,128] @ [128,128] per batch)
// grid (32, 8), block 256 (8 waves). bf16 WMMA, f32 accumulate.
// ---------------------------------------------------------------------------
__global__ __launch_bounds__(256) void proj_kernel(const float* __restrict__ X,
                                                   const float* __restrict__ W,
                                                   float* __restrict__ Y) {
  __shared__ __align__(16) bf16_t Wt[128 * 136];  // W^T: [out][in]
  __shared__ __align__(16) bf16_t Xs[64 * 136];   // [n][in]
  const int b = blockIdx.y, n0 = blockIdx.x * 64;
  const int t = threadIdx.x;

  for (int i = 0; i < 64; ++i) {                  // 128*128 / 256
    int idx = t + i * 256;
    int r = idx >> 7, c = idx & 127;              // r = in, c = out
    Wt[c * 136 + r] = (bf16_t)W[idx];
  }
  const float* Xb = X + ((size_t)b * NS + n0) * ND;
  for (int i = 0; i < 32; ++i) {                  // 64*128 / 256
    int idx = t + i * 256;
    int r = idx >> 7, c = idx & 127;
    Xs[r * 136 + c] = (bf16_t)Xb[idx];
  }
  __syncthreads();

  const int w = t >> 5, lane = t & 31;
  const int rowsub = w >> 1, colh = w & 1;

  v16bf xa[4];
#pragma unroll
  for (int ks = 0; ks < 4; ++ks)
    xa[ks] = load_frag_a(Xs + rowsub * 16 * 136 + ks * 32, 136, lane);

  float* Yb = Y + ((size_t)b * NS + n0) * ND;
  const int rbase = rowsub * 16 + 8 * (lane >> 4);
#pragma unroll
  for (int cs = 0; cs < 4; ++cs) {
    int col0 = colh * 64 + cs * 16;
    v8f acc = {0.f, 0.f, 0.f, 0.f, 0.f, 0.f, 0.f, 0.f};
#pragma unroll
    for (int ks = 0; ks < 4; ++ks)
      acc = wmma_bf16(xa[ks], load_frag_b(Wt + col0 * 136 + ks * 32, 136, lane), acc);
    int ncol = col0 + (lane & 15);
#pragma unroll
    for (int r = 0; r < 8; ++r)
      Yb[(size_t)(rbase + r) * ND + ncol] = acc[r];
  }
}

// ---------------------------------------------------------------------------
// Stage B1: P[b,m,d] = prefix_sum_m( gamma^{-m} * colscale[m] * K[b,m,d] )
// One wave per (b,d) column; 64 chunks of 32 with shuffle scan + carry.
// grid 256, block 128 (4 waves).
// ---------------------------------------------------------------------------
__global__ __launch_bounds__(128) void scan_kernel(const float* __restrict__ K,
                                                   float* __restrict__ P) {
  const int wid = threadIdx.x >> 5, lane = threadIdx.x & 31;
  const int cid = blockIdx.x * 4 + wid;           // 0..1023
  const int b = cid >> 7, d = cid & 127;
  const float LG = __logf(GAMMA_F);
  const float* Kb = K + (size_t)b * NS * ND + d;
  float* Pb = P + (size_t)b * NS * ND + d;
  float carry = 0.f;
  for (int s = 0; s < 64; ++s) {
    int m = s * 32 + lane;
    float csc = sqrtf((1.f - GAMMA_F) / (1.f - __expf((float)(m + 1) * LG)));
    float v = Kb[(size_t)m * ND] * csc * __expf(-(float)m * LG);
#pragma unroll
    for (int off = 1; off < 32; off <<= 1) {
      float tmp = __shfl_up(v, off, 32);
      if (lane >= off) v += tmp;
    }
    v += carry;
    Pb[(size_t)m * ND] = v;
    carry = __shfl(v, 31, 32);
  }
}

// ---------------------------------------------------------------------------
// Stage B2: s[b,n] = gamma^n / sqrt(D) * dot(Q[b,n,:], P[b,n,:])
//           nrm[b,n] = colscale[n] / max(|s|, 1)
// One wave per row. grid 2048, block 256 (8 waves).
// ---------------------------------------------------------------------------
__global__ __launch_bounds__(256) void norm_kernel(const float* __restrict__ Q,
                                                   const float* __restrict__ P,
                                                   float* __restrict__ nrm) {
  const int wid = threadIdx.x >> 5, lane = threadIdx.x & 31;
  const int row = blockIdx.x * 8 + wid;           // 0..16383 == b*2048+n
  const int n = row & (NS - 1);
  const float* q = Q + (size_t)row * ND;
  const float* p = P + (size_t)row * ND;
  float acc = 0.f;
#pragma unroll
  for (int i = 0; i < 4; ++i) acc += q[lane + 32 * i] * p[lane + 32 * i];
#pragma unroll
  for (int off = 16; off > 0; off >>= 1) acc += __shfl_xor(acc, off, 32);
  const float LG = __logf(GAMMA_F);
  float s = __expf((float)n * LG) * RSQRT_D * acc;
  if (lane == 0) {
    float csc = sqrtf((1.f - GAMMA_F) / (1.f - __expf((float)(n + 1) * LG)));
    nrm[row] = csc / fmaxf(fabsf(s), 1.f);
  }
}

// ---------------------------------------------------------------------------
// Stage C: flash-style retention output.
// out[n,:] = sum_{m<=n} gamma^{n-m} * nrm[m] * (Q[n]·K[m]/sqrt(D)) * V[m,:]
// grid (32, 8), block 256. Decay folded: gamma^{n-n0} into Q (once),
// gamma^{m0-m}*nrm[m] into K (per tile), scalar gamma^{n0-m0} per tile.
// ---------------------------------------------------------------------------
__global__ __launch_bounds__(256) void retention_kernel(const float* __restrict__ Q,
                                                        const float* __restrict__ K,
                                                        const float* __restrict__ V,
                                                        const float* __restrict__ nrm,
                                                        float* __restrict__ out) {
  __shared__ __align__(16) bf16_t Qs[64 * 136];    // [n][d], pre-scaled
  __shared__ __align__(16) bf16_t Ks[32 * 136];    // [m][d], pre-scaled
  __shared__ __align__(16) bf16_t Vs[128 * 40];    // V^T: [d][m]
  __shared__ __align__(16) bf16_t Ss[8 * 16 * 40]; // per-wave score tile [16][32]
  const int b = blockIdx.y, n0 = blockIdx.x * 64;
  const int t = threadIdx.x, w = t >> 5, lane = t & 31;
  const float LG = __logf(GAMMA_F);

  const float* Qb = Q + ((size_t)b * NS + n0) * ND;
  for (int i = 0; i < 32; ++i) {                   // 64*128 / 256
    int idx = t + i * 256;
    int r = idx >> 7, c = idx & 127;
    float sc = __expf((float)r * LG) * RSQRT_D;    // gamma^{n-n0} / sqrt(D)
    Qs[r * 136 + c] = (bf16_t)(Qb[idx] * sc);
  }
  __syncthreads();

  const int rowsub = w >> 1, colh = w & 1;
  const int h = lane >> 4, nl = lane & 15;

  v16bf qa[4];
#pragma unroll
  for (int ks = 0; ks < 4; ++ks)
    qa[ks] = load_frag_a(Qs + rowsub * 16 * 136 + ks * 32, 136, lane);

  v8f oacc[4];
#pragma unroll
  for (int cs = 0; cs < 4; ++cs)
#pragma unroll
    for (int r = 0; r < 8; ++r) oacc[cs][r] = 0.f;

  for (int m0 = 0; m0 < n0 + 64; m0 += 32) {
    __syncthreads();  // previous tile fully consumed
    const float* Kb = K + ((size_t)b * NS + m0) * ND;
    const float* Vb = V + ((size_t)b * NS + m0) * ND;
    const float* nb = nrm + (size_t)b * NS + m0;
    for (int i = 0; i < 16; ++i) {                 // 32*128 / 256
      int idx = t + i * 256;
      int r = idx >> 7, c = idx & 127;             // r = m-local, c = d
      float f = nb[r] * __expf(-(float)r * LG);    // nrm[m] * gamma^{m0-m}
      Ks[r * 136 + c] = (bf16_t)(Kb[idx] * f);
      Vs[c * 40 + r] = (bf16_t)Vb[idx];
    }
    __syncthreads();

    // scores: 16x32 per wave, f32 accumulate
    v8f sacc[2];
#pragma unroll
    for (int ms = 0; ms < 2; ++ms)
#pragma unroll
      for (int r = 0; r < 8; ++r) sacc[ms][r] = 0.f;
#pragma unroll
    for (int ms = 0; ms < 2; ++ms)
#pragma unroll
      for (int ks = 0; ks < 4; ++ks)
        sacc[ms] = wmma_bf16(qa[ks],
                             load_frag_b(Ks + ms * 16 * 136 + ks * 32, 136, lane),
                             sacc[ms]);

    const float tf = __expf((float)(n0 - m0) * LG);  // gamma^{n0-m0}
    bf16_t* Sw = Ss + w * 16 * 40;
#pragma unroll
    for (int ms = 0; ms < 2; ++ms) {
#pragma unroll
      for (int r = 0; r < 8; ++r) {
        float v = sacc[ms][r] * tf;
        if (m0 >= n0) {                              // diagonal tiles: causal mask
          int nn = n0 + rowsub * 16 + r + 8 * h;
          int mm = m0 + ms * 16 + nl;
          if (mm > nn) v = 0.f;
        }
        Sw[(r + 8 * h) * 40 + ms * 16 + nl] = (bf16_t)v;
      }
    }
    __syncthreads();

    // out += S @ V   (A = 16x32 scores, B = V^T column tiles)
    v16bf sa = load_frag_a(Sw, 40, lane);
#pragma unroll
    for (int cs = 0; cs < 4; ++cs) {
      int col0 = colh * 64 + cs * 16;
      oacc[cs] = wmma_bf16(sa, load_frag_b(Vs + col0 * 40, 40, lane), oacc[cs]);
    }
  }

  float* Ob = out + ((size_t)b * NS + n0) * ND;
  const int rbase = rowsub * 16 + 8 * h;
#pragma unroll
  for (int cs = 0; cs < 4; ++cs) {
    int ncol = colh * 64 + cs * 16 + nl;
#pragma unroll
    for (int r = 0; r < 8; ++r)
      Ob[(size_t)(rbase + r) * ND + ncol] = oacc[cs][r];
  }
}

// ---------------------------------------------------------------------------
extern "C" void kernel_launch(void* const* d_in, const int* in_sizes, int n_in,
                              void* d_out, int out_size, void* d_ws, size_t ws_size,
                              hipStream_t stream) {
  const float* xq = (const float*)d_in[0];
  const float* xk = (const float*)d_in[1];
  const float* xv = (const float*)d_in[2];
  const float* W  = (const float*)d_in[3];
  float* out = (float*)d_out;

  float* ws = (float*)d_ws;
  float* Q   = ws;                 // BSD floats
  float* K   = ws + (size_t)BSD;
  float* V   = ws + (size_t)2 * BSD;
  float* P   = ws + (size_t)3 * BSD;
  float* nrm = ws + (size_t)4 * BSD;  // NB*NS floats

  dim3 gproj(NS / 64, NB);
  proj_kernel<<<gproj, 256, 0, stream>>>(xq, W, Q);
  proj_kernel<<<gproj, 256, 0, stream>>>(xk, W, K);
  proj_kernel<<<gproj, 256, 0, stream>>>(xv, W, V);

  scan_kernel<<<256, 128, 0, stream>>>(K, P);
  norm_kernel<<<2048, 256, 0, stream>>>(Q, P, nrm);

  dim3 gret(NS / 64, NB);
  retention_kernel<<<gret, 256, 0, stream>>>(Q, K, V, nrm, out);
}
